// Attention_52278341927646
// MI455X (gfx1250) — matrix operand order repaired and checked by reading, not compiled, not run on hardware
//
#include <hip/hip_runtime.h>
#include <stdint.h>

#define ENC 2048
#define DEC 512
#define ATT 512
#define BB  128
#define LL  196

typedef __attribute__((ext_vector_type(16))) __bf16 v16bf;
typedef __attribute__((ext_vector_type(8)))  float  v8f;

union Frag16 { uint4 u[2]; v16bf v; };

__device__ __forceinline__ unsigned short f2bf(float x) {
  unsigned int u = __float_as_uint(x);
  unsigned int r = u + 0x7fffu + ((u >> 16) & 1u);   // round-to-nearest-even
  return (unsigned short)(r >> 16);
}

// ---------------------------------------------------------------------------
// Kernel 0: repack U_w [2048x512] fp32 -> bf16 in WMMA B-fragment order.
// Layout: Bp[kc][ntile][lane][e], kc=K/32 chunk, ntile=N/16, lane 0..31,
// e 0..15 with K = kc*32 + (lane>>4)*16 + e, N = ntile*16 + (lane&15).
// Each lane's fragment is 32 contiguous bytes -> two coalesced b128 loads.
// ---------------------------------------------------------------------------
__global__ void pack_Uw_bf16(const float* __restrict__ Uw,
                             unsigned short* __restrict__ Bp) {
  int id = blockIdx.x * blockDim.x + threadIdx.x;   // 0..65535
  int lane  = id & 31;
  int ntile = (id >> 5) & 31;
  int kc    = id >> 10;                              // 0..63
  int n     = ntile * 16 + (lane & 15);
  int kbase = kc * 32 + (lane >> 4) * 16;
  unsigned int pk[8];
#pragma unroll
  for (int e = 0; e < 8; ++e) {
    unsigned short lo = f2bf(Uw[(size_t)(kbase + 2 * e) * ATT + n]);
    unsigned short hi = f2bf(Uw[(size_t)(kbase + 2 * e + 1) * ATT + n]);
    pk[e] = (unsigned int)lo | ((unsigned int)hi << 16);
  }
  uint4* dst = (uint4*)(Bp + (size_t)id * 16);
  dst[0] = make_uint4(pk[0], pk[1], pk[2], pk[3]);
  dst[1] = make_uint4(pk[4], pk[5], pk[6], pk[7]);
}

// ---------------------------------------------------------------------------
// Kernel 1: w_ah[b][a] = hidden[b,:] @ W_w[:,a] + W_b[a]   (tiny, VALU)
// ---------------------------------------------------------------------------
__global__ void __launch_bounds__(256)
wah_kernel(const float* __restrict__ hidden, const float* __restrict__ Ww,
           const float* __restrict__ Wb, float* __restrict__ wah) {
  __shared__ float hs[DEC];
  const int b = blockIdx.x, t = threadIdx.x;
  hs[t]       = hidden[b * DEC + t];
  hs[t + 256] = hidden[b * DEC + t + 256];
  __syncthreads();
  float a0 = Wb[t], a1 = Wb[t + 256];
  for (int d = 0; d < DEC; ++d) {
    float h = hs[d];
    a0 = fmaf(h, Ww[(size_t)d * ATT + t], a0);
    a1 = fmaf(h, Ww[(size_t)d * ATT + t + 256], a1);
  }
  wah[b * ATT + t]       = a0;
  wah[b * ATT + t + 256] = a1;
}

// ---------------------------------------------------------------------------
// Kernel 2: scores[b,l] = A_w . tanh(features[b,l,:]@U_w + U_b + w_ah[b,:]) + A_b
// One workgroup = (b, 32 rows of L). 8 waves = 2 M-tiles x 4 N-groups.
// bf16 WMMA 16x16x32, f32 accumulate, K = 2048 in 64 chunks.
// ---------------------------------------------------------------------------
__global__ void __launch_bounds__(256)
scores_kernel(const float* __restrict__ features,
              const unsigned short* __restrict__ Bp,
              const float* __restrict__ Ub, const float* __restrict__ wah,
              const float* __restrict__ Aw, const float* __restrict__ Ab,
              float* __restrict__ scores) {
  extern __shared__ char smem[];
  unsigned short* ldsA = (unsigned short*)smem;                       // [32][ENC] bf16
  float* sScore = (float*)(smem + (size_t)32 * ENC * sizeof(unsigned short)); // [32]

  const int b    = blockIdx.y;
  const int l0   = blockIdx.x * 32;
  const int tid  = threadIdx.x;
  const int lane = tid & 31;
  const int wave = tid >> 5;
  const int mt   = wave >> 2;   // 0..1 : M-tile
  const int ng   = wave & 3;    // 0..3 : N-group of 128
  const int ln   = lane & 15;
  const int kh   = lane >> 4;

  if (tid < 32) sScore[tid] = 0.0f;

  // Stage A tile: 32 rows x 2048, fp32 -> bf16 row-major LDS (zero-pad l>=196).
#pragma unroll 4
  for (int i = 0; i < 64; ++i) {
    int c   = tid + i * 256;     // float4 chunk id, 512 chunks per row
    int row = c >> 9;
    int c4  = c & 511;
    int l   = l0 + row;
    float4 f = make_float4(0.f, 0.f, 0.f, 0.f);
    if (l < LL)
      f = *(const float4*)(features + ((size_t)(b * LL + l) * ENC + c4 * 4));
    unsigned int p0 = (unsigned int)f2bf(f.x) | ((unsigned int)f2bf(f.y) << 16);
    unsigned int p1 = (unsigned int)f2bf(f.z) | ((unsigned int)f2bf(f.w) << 16);
    *(uint2*)(ldsA + (size_t)row * ENC + c4 * 4) = make_uint2(p0, p1);
  }
  __syncthreads();

  v8f zero = {0.f, 0.f, 0.f, 0.f, 0.f, 0.f, 0.f, 0.f};
  v8f acc[8];
#pragma unroll
  for (int j = 0; j < 8; ++j) acc[j] = zero;

  // A-fragment base: row = mt*16 + ln, K elements 0-15 at kh*16 (contiguous).
  const unsigned short* arow = ldsA + (size_t)(mt * 16 + ln) * ENC + kh * 16;

  for (int kc = 0; kc < 64; ++kc) {
    Frag16 a;
    a.u[0] = *(const uint4*)(arow + kc * 32);
    a.u[1] = *(const uint4*)(arow + kc * 32 + 8);
    const unsigned short* bpk = Bp + (((size_t)kc * 32 + ng * 8) * 32 + lane) * 16;
#pragma unroll
    for (int j = 0; j < 8; ++j) {
      Frag16 bf;
      const uint4* bp4 = (const uint4*)(bpk + (size_t)j * 32 * 16);
      bf.u[0] = bp4[0];
      bf.u[1] = bp4[1];
      acc[j] = __builtin_amdgcn_wmma_f32_16x16x32_bf16(
          false, a.v, false, bf.v, (short)0, acc[j], false, false);
    }
    // prefetch next K-chunk of packed B (L2-resident, keep it warm)
    if (kc + 1 < 64)
      __builtin_prefetch(bpk + (size_t)32 * 32 * 16, 0, 1);
  }

  // Epilogue: C layout -> m = r + 8*kh, n = lane&15 (per N-tile).
  float partial[8];
#pragma unroll
  for (int r = 0; r < 8; ++r) partial[r] = 0.0f;
#pragma unroll
  for (int j = 0; j < 8; ++j) {
    int n = ng * 128 + j * 16 + ln;
    float bias = Ub[n] + wah[b * ATT + n];
    float awn  = Aw[n];
#pragma unroll
    for (int r = 0; r < 8; ++r) {
      float v = acc[j][r] + bias;
      partial[r] += tanhf(v) * awn;
    }
  }
#pragma unroll
  for (int r = 0; r < 8; ++r)
    atomicAdd(&sScore[mt * 16 + kh * 8 + r], partial[r]);   // ds_add_f32
  __syncthreads();

  if (tid < 32) {
    int l = l0 + tid;
    if (l < LL) scores[b * LL + l] = sScore[tid] + Ab[0];
  }
}

// ---------------------------------------------------------------------------
// Kernel 3: softmax over L per b, write alpha, then context = alpha @ features.
// Fully coalesced: one block reads one full 8KB feature row per l iteration.
// ---------------------------------------------------------------------------
__global__ void __launch_bounds__(256)
softmax_ctx_kernel(const float* __restrict__ features,
                   const float* __restrict__ scores,
                   float* __restrict__ alpha_out, float* __restrict__ ctx_out) {
  __shared__ float red[256];
  __shared__ float sAlpha[LL];
  const int b = blockIdx.x, t = threadIdx.x;

  float s = (t < LL) ? scores[b * LL + t] : -3.0e38f;
  red[t] = s;
  __syncthreads();
  for (int off = 128; off > 0; off >>= 1) {
    if (t < off) red[t] = fmaxf(red[t], red[t + off]);
    __syncthreads();
  }
  float mx = red[0];
  __syncthreads();
  float e = (t < LL) ? __expf(s - mx) : 0.0f;
  red[t] = e;
  __syncthreads();
  for (int off = 128; off > 0; off >>= 1) {
    if (t < off) red[t] += red[t + off];
    __syncthreads();
  }
  float a = e * (1.0f / red[0]);
  if (t < LL) {
    alpha_out[b * LL + t] = a;
    sAlpha[t] = a;
  }
  __syncthreads();

  const float* fb = features + (size_t)b * LL * ENC + t * 8;
  float c[8];
#pragma unroll
  for (int i = 0; i < 8; ++i) c[i] = 0.0f;
  for (int l = 0; l < LL; ++l) {
    float  al = sAlpha[l];
    float4 f0 = *(const float4*)(fb + (size_t)l * ENC);
    float4 f1 = *(const float4*)(fb + (size_t)l * ENC + 4);
    c[0] = fmaf(al, f0.x, c[0]); c[1] = fmaf(al, f0.y, c[1]);
    c[2] = fmaf(al, f0.z, c[2]); c[3] = fmaf(al, f0.w, c[3]);
    c[4] = fmaf(al, f1.x, c[4]); c[5] = fmaf(al, f1.y, c[5]);
    c[6] = fmaf(al, f1.z, c[6]); c[7] = fmaf(al, f1.w, c[7]);
  }
  float* dst = ctx_out + (size_t)b * ENC + t * 8;
  *(float4*)dst       = make_float4(c[0], c[1], c[2], c[3]);
  *((float4*)dst + 1) = make_float4(c[4], c[5], c[6], c[7]);
}

// ---------------------------------------------------------------------------
// Workspace layout (needs ~2.62 MB):
//   [0,       256KB) w_ah      128x512 f32
//   [256KB,   356KB) scores    128x196 f32
//   [512KB,  2560KB) Bp        packed bf16 U_w (64x32x32x16 ushort = 2MB)
// ---------------------------------------------------------------------------
extern "C" void kernel_launch(void* const* d_in, const int* in_sizes, int n_in,
                              void* d_out, int out_size, void* d_ws, size_t ws_size,
                              hipStream_t stream) {
  const float* features = (const float*)d_in[0];
  const float* hidden   = (const float*)d_in[1];
  const float* Uw       = (const float*)d_in[2];
  const float* Ub       = (const float*)d_in[3];
  const float* Ww       = (const float*)d_in[4];
  const float* Wb       = (const float*)d_in[5];
  const float* Aw       = (const float*)d_in[6];
  const float* Ab       = (const float*)d_in[7];
  float* out = (float*)d_out;

  char* ws = (char*)d_ws;
  float* wah            = (float*)(ws);
  float* scores         = (float*)(ws + 262144);
  unsigned short* Bp    = (unsigned short*)(ws + 524288);

  pack_Uw_bf16<<<256, 256, 0, stream>>>(Uw, Bp);
  wah_kernel<<<BB, 256, 0, stream>>>(hidden, Ww, Wb, wah);

  size_t smem2 = (size_t)32 * ENC * sizeof(unsigned short) + 32 * sizeof(float);
  scores_kernel<<<dim3(7, BB), 256, smem2, stream>>>(features, Bp, Ub, wah, Aw, Ab, scores);

  softmax_ctx_kernel<<<BB, 256, 0, stream>>>(features, scores, out, out + BB * LL);
}
